// BertSelfAttentionNoAttMask_63230508532014
// MI455X (gfx1250) — matrix operand order repaired and checked by reading, not compiled, not run on hardware
//
#include <hip/hip_runtime.h>

// Problem constants (BERT self-attention, no mask)
#define Hdim   1024
#define NHEAD  16
#define HDIM   64
#define BATCH  4
#define SEQ    2048
#define BS     (BATCH * SEQ)   // 8192 tokens

typedef __attribute__((ext_vector_type(16))) _Float16 v16h;
typedef __attribute__((ext_vector_type(8)))  _Float16 v8h;
typedef __attribute__((ext_vector_type(8)))  float    v8f;

// ---------------------------------------------------------------------------
// WMMA helpers (CDNA5 16x16x32 f16 -> f32)
// ---------------------------------------------------------------------------
static __device__ __forceinline__ v8f wmma_f16(v16h a, v16h b, v8f c) {
  // (neg_a, A, neg_b, B, c_mod, C, reuse_a, reuse_b)
  return __builtin_amdgcn_wmma_f32_16x16x32_f16(false, a, false, b, (short)0, c,
                                                false, false);
}

// A-matrix 16x32 f16 fragment (contiguous K, matches load_bt):
//   lane L<16 : row=L,    K = k0..k0+15, k0=0
//   lane L>=16: row=L-16, same with k0=16
// Caller passes p = base + row*ld + k0 ; chunks at p and p+8.
static __device__ __forceinline__ v16h load_a16(const _Float16* p) {
  v8h lo = *(const v8h*)p;
  v8h hi = *(const v8h*)(p + 8);
  v16h a;
#pragma unroll
  for (int i = 0; i < 8; ++i) { a[i] = lo[i]; a[i + 8] = hi[i]; }
  return a;
}

// B-matrix 32x16 f16 fragment from B-transposed storage (rows = N, contiguous K):
//   lane L<16 : N=L, K=0..15 ; lane L>=16: N=L-16, K=16..31
// Caller passes p = bt + n*ld + kb  (kb = (lane>=16)*16) -> one 32B load.
static __device__ __forceinline__ v16h load_bt(const _Float16* p) {
  return *(const v16h*)p;
}

// Half-wave (16-lane) max reduction: C/D layout keeps one matrix row in 16 lanes.
static __device__ __forceinline__ float hmax16(float v) {
#pragma unroll
  for (int m = 1; m <= 8; m <<= 1) v = fmaxf(v, __shfl_xor(v, m, 32));
  return v;
}

// ---------------------------------------------------------------------------
// f32 -> f16 conversion kernels
// ---------------------------------------------------------------------------
__global__ __launch_bounds__(256) void cvt_f32_f16(const float* __restrict__ src,
                                                   _Float16* __restrict__ dst,
                                                   int n) {
  int i = blockIdx.x * blockDim.x + threadIdx.x;
  int stride = gridDim.x * blockDim.x;
  for (; i < n; i += stride) dst[i] = (_Float16)src[i];
}

// W[k][n] (1024x1024) -> Wt[n][k] f16 (so GEMM B-fragments are contiguous)
__global__ __launch_bounds__(256) void cvt_w_t(const float* __restrict__ W,
                                               _Float16* __restrict__ Wt) {
  int i = blockIdx.x * blockDim.x + threadIdx.x;
  if (i >= Hdim * Hdim) return;
  int n = i & (Hdim - 1);
  int k = i >> 10;
  Wt[n * Hdim + k] = (_Float16)W[k * Hdim + n];
}

// ---------------------------------------------------------------------------
// QKV projection: Y = Xh[8192x1024] @ Wt^T + bias, per-wave 16x64 output tile.
// Waves of a block share tileN (same B fragments -> WGP$ reuse), consecutive tileM.
// Software-pipelined k-loop: next A/B fragments loaded before current WMMAs.
//   transposeV==0: out f16 [b][h][s][d]   (Q uses scale = 0.125*log2e, K scale=1)
//   transposeV==1: out f16 [b][h][d][s]   (V transposed for PV B-fragments)
// ---------------------------------------------------------------------------
__global__ __launch_bounds__(256) void qkv_proj(const _Float16* __restrict__ Xh,
                                                const _Float16* __restrict__ Wt,
                                                const float* __restrict__ bias,
                                                _Float16* __restrict__ out,
                                                float scale, int transposeV) {
  const int lane  = threadIdx.x & 31;
  const int wave  = blockIdx.x * 8 + (threadIdx.x >> 5);
  const int tileM = wave & 511;       // 0..511 (16 rows each); block shares tileN
  const int tileN = wave >> 9;        // head index (64 cols each)
  const int row16 = tileM * 16;

  const int arow = lane & 15;
  const int ak0  = (lane >> 4) * 16;
  const int bn   = lane & 15;
  const int bk   = (lane >> 4) * 16;

  const _Float16* aptr = Xh + (size_t)(row16 + arow) * Hdim + ak0;
  const _Float16* bptr = Wt + (size_t)(tileN * HDIM + bn) * Hdim + bk;

  v8f acc[4] = {};

  // Pipelined main loop: fragments for step kt live while kt+32 loads are in flight.
  v16h a  = load_a16(aptr);
  v16h b0 = load_bt(bptr + 0 * 16 * Hdim);
  v16h b1 = load_bt(bptr + 1 * 16 * Hdim);
  v16h b2 = load_bt(bptr + 2 * 16 * Hdim);
  v16h b3 = load_bt(bptr + 3 * 16 * Hdim);

  for (int kt = 0; kt < Hdim - 32; kt += 32) {
    v16h an  = load_a16(aptr + kt + 32);
    v16h nb0 = load_bt(bptr + 0 * 16 * Hdim + kt + 32);
    v16h nb1 = load_bt(bptr + 1 * 16 * Hdim + kt + 32);
    v16h nb2 = load_bt(bptr + 2 * 16 * Hdim + kt + 32);
    v16h nb3 = load_bt(bptr + 3 * 16 * Hdim + kt + 32);
    acc[0] = wmma_f16(a, b0, acc[0]);
    acc[1] = wmma_f16(a, b1, acc[1]);
    acc[2] = wmma_f16(a, b2, acc[2]);
    acc[3] = wmma_f16(a, b3, acc[3]);
    a = an; b0 = nb0; b1 = nb1; b2 = nb2; b3 = nb3;
  }
  acc[0] = wmma_f16(a, b0, acc[0]);
  acc[1] = wmma_f16(a, b1, acc[1]);
  acc[2] = wmma_f16(a, b2, acc[2]);
  acc[3] = wmma_f16(a, b3, acc[3]);

  const int hi = lane >> 4;   // C layout: VGPR r -> row r + 8*hi, col = lane&15
  const int n  = lane & 15;
  const int bb = row16 >> 11; // batch (16-row tile never crosses batch boundary)

  if (!transposeV) {
#pragma unroll
    for (int t = 0; t < 4; ++t) {
      float bc = bias[tileN * HDIM + t * 16 + n];
#pragma unroll
      for (int r = 0; r < 8; ++r) {
        int m = row16 + r + 8 * hi;
        int s = m & (SEQ - 1);
        float v = (acc[t][r] + bc) * scale;
        out[(((size_t)(bb * NHEAD + tileN) * SEQ + s) * HDIM) + t * 16 + n] =
            (_Float16)v;
      }
    }
  } else {
    // Transposed store: per lane, rows r..r+7 are consecutive s -> packed v8h
#pragma unroll
    for (int t = 0; t < 4; ++t) {
      float bc = bias[tileN * HDIM + t * 16 + n];
      v8h pk;
#pragma unroll
      for (int r = 0; r < 8; ++r) pk[r] = (_Float16)((acc[t][r] + bc) * scale);
      int d  = t * 16 + n;
      int s0 = (row16 + 8 * hi) & (SEQ - 1);
      *(v8h*)&out[((size_t)(bb * NHEAD + tileN) * HDIM + d) * SEQ + s0] = pk;
    }
  }
}

// ---------------------------------------------------------------------------
// Flash attention: one wave per (b,h, 16-query tile); key tiles of 32.
// Q pre-scaled by 0.125*log2e so softmax = exp2(c - rowmax) (invariant).
// All 8 waves of a block share one (b,h) -> K/V tiles reused via WGP$/L2.
// Softmax denominator computed on the matrix pipe via a "ones" B-tile
// (l = P @ 1 lands in col 0 of a 5th accumulator, alpha-corrected like ctx).
// K fragments for jt+1 prefetched during softmax; V fragments issued early.
// ---------------------------------------------------------------------------
__global__ __launch_bounds__(256) void attn_kernel(const _Float16* __restrict__ Qh,
                                                   const _Float16* __restrict__ Kh,
                                                   const _Float16* __restrict__ Vt,
                                                   float* __restrict__ out) {
  __shared__ _Float16 plds[8 * 16 * 40];  // per-wave 16x32 P tile, stride 40

  const int lane = threadIdx.x & 31;
  const int widx = threadIdx.x >> 5;
  const int bh   = blockIdx.x >> 4;                 // 0..63
  const int qt   = (blockIdx.x & 15) * 8 + widx;    // 0..127

  const _Float16* Qb = Qh + (size_t)bh * SEQ * HDIM + (size_t)qt * 16 * HDIM;
  const _Float16* Kb = Kh + (size_t)bh * SEQ * HDIM;
  const _Float16* Vb = Vt + (size_t)bh * HDIM * SEQ;
  _Float16* P = plds + widx * (16 * 40);

  const int arow = lane & 15;
  const int ak0  = (lane >> 4) * 16;
  const int bn   = lane & 15;
  const int bk   = (lane >> 4) * 16;

  // Q A-fragments for K-windows [0,32) and [32,64), kept resident all loop.
  v16h qa0 = load_a16(Qb + arow * HDIM + ak0);
  v16h qa1 = load_a16(Qb + arow * HDIM + 32 + ak0);

  // "ones" B-tile: column 0 of B is all ones -> D col 0 = row-sum of A.
  v16h ones;
  {
    _Float16 ov = (_Float16)((lane & 15) == 0 ? 1.0f : 0.0f);
#pragma unroll
    for (int i = 0; i < 16; ++i) ones[i] = ov;
  }

  float m[8];
#pragma unroll
  for (int r = 0; r < 8; ++r) m[r] = -INFINITY;
  v8f acc[4] = {};
  v8f accl = {};  // softmax denominator accumulator (col 0)

  // Preload K fragments for jt = 0.
  v16h kb0 = load_bt(Kb + bn * HDIM + bk);
  v16h kb1 = load_bt(Kb + bn * HDIM + 32 + bk);
  v16h kb2 = load_bt(Kb + (16 + bn) * HDIM + bk);
  v16h kb3 = load_bt(Kb + (16 + bn) * HDIM + 32 + bk);

  for (int jt = 0; jt < SEQ / 32; ++jt) {
    // V fragments for this iteration, issued early (hidden behind score WMMAs).
    v16h vb0 = load_bt(Vb + (size_t)(0 * 16 + bn) * SEQ + jt * 32 + bk);
    v16h vb1 = load_bt(Vb + (size_t)(1 * 16 + bn) * SEQ + jt * 32 + bk);
    v16h vb2 = load_bt(Vb + (size_t)(2 * 16 + bn) * SEQ + jt * 32 + bk);
    v16h vb3 = load_bt(Vb + (size_t)(3 * 16 + bn) * SEQ + jt * 32 + bk);

    // scores tile 16x32 = two 16x16 C tiles, head-dim K=64 -> 2 WMMAs each
    v8f c0 = {}, c1 = {};
    c0 = wmma_f16(qa0, kb0, c0);
    c0 = wmma_f16(qa1, kb1, c0);
    c1 = wmma_f16(qa0, kb2, c1);
    c1 = wmma_f16(qa1, kb3, c1);

    // Prefetch K fragments for jt+1 (clamped on last iteration, branchless).
    int nj = jt + 1 < SEQ / 32 ? jt + 1 : jt;
    const _Float16* Kn = Kb + (size_t)(nj * 32) * HDIM;
    v16h nk0 = load_bt(Kn + bn * HDIM + bk);
    v16h nk1 = load_bt(Kn + bn * HDIM + 32 + bk);
    v16h nk2 = load_bt(Kn + (16 + bn) * HDIM + bk);
    v16h nk3 = load_bt(Kn + (16 + bn) * HDIM + 32 + bk);

    // Streaming softmax (row-max reductions across the 16-lane half groups).
    float nm[8], al[8];
#pragma unroll
    for (int r = 0; r < 8; ++r) {
      float t = hmax16(fmaxf(c0[r], c1[r]));
      nm[r] = fmaxf(m[r], t);
      al[r] = exp2f(m[r] - nm[r]);   // first iter: exp2(-inf) = 0
      m[r]  = nm[r];
    }

    const int prow = (lane >> 4) * 8;
#pragma unroll
    for (int r = 0; r < 8; ++r) {
      float e0 = exp2f(c0[r] - nm[r]);
      float e1 = exp2f(c1[r] - nm[r]);
      int row = prow + r;
      P[row * 40 + (lane & 15)]      = (_Float16)e0;
      P[row * 40 + 16 + (lane & 15)] = (_Float16)e1;
    }
    asm volatile("s_wait_dscnt 0" ::: "memory");

    // Reload P in A-fragment layout from LDS (16B-aligned ds_load_b128s)
    v16h pa = load_a16(P + arow * 40 + ak0);

    // l accumulator: alpha-correct then l += P @ ones (matrix pipe, no shuffles)
    {
      v8f a5 = accl;
#pragma unroll
      for (int r = 0; r < 8; ++r) a5[r] *= al[r];
      accl = wmma_f16(pa, ones, a5);
    }
#pragma unroll
    for (int t = 0; t < 4; ++t) {
      v8f a = acc[t];
#pragma unroll
      for (int r = 0; r < 8; ++r) a[r] *= al[r];
      v16h vb = (t == 0) ? vb0 : (t == 1) ? vb1 : (t == 2) ? vb2 : vb3;
      acc[t] = wmma_f16(pa, vb, a);
    }

    kb0 = nk0; kb1 = nk1; kb2 = nk2; kb3 = nk3;
  }

  // Normalize and store fp32 output [b][s][h*64+d].
  // l for row (r + 8*hi) lives in col 0 of accl: lane 0 (rows 0-7) / lane 16 (rows 8-15).
  const int b = bh >> 4, h = bh & 15;
  const int hi = lane >> 4, n = lane & 15;
  float inv[8];
#pragma unroll
  for (int r = 0; r < 8; ++r) {
    float lv = __shfl(accl[r], lane & 16, 32);
    inv[r] = 1.0f / lv;
  }
#pragma unroll
  for (int t = 0; t < 4; ++t) {
#pragma unroll
    for (int r = 0; r < 8; ++r) {
      int s = qt * 16 + r + 8 * hi;
      int col = h * HDIM + t * 16 + n;
      out[((size_t)(b * SEQ + s)) * Hdim + col] = acc[t][r] * inv[r];
    }
  }
}

// ---------------------------------------------------------------------------
// Launch
// ---------------------------------------------------------------------------
extern "C" void kernel_launch(void* const* d_in, const int* in_sizes, int n_in,
                              void* d_out, int out_size, void* d_ws, size_t ws_size,
                              hipStream_t stream) {
  const float* X  = (const float*)d_in[0];
  const float* Wq = (const float*)d_in[1];
  const float* bq = (const float*)d_in[2];
  const float* Wk = (const float*)d_in[3];
  const float* bk = (const float*)d_in[4];
  const float* Wv = (const float*)d_in[5];
  const float* bv = (const float*)d_in[6];
  float* out = (float*)d_out;

  char* w = (char*)d_ws;
  _Float16* Xh  = (_Float16*)w; w += (size_t)BS * Hdim * 2;      // 16 MB
  _Float16* Wtq = (_Float16*)w; w += (size_t)Hdim * Hdim * 2;    // 2 MB
  _Float16* Wtk = (_Float16*)w; w += (size_t)Hdim * Hdim * 2;
  _Float16* Wtv = (_Float16*)w; w += (size_t)Hdim * Hdim * 2;
  _Float16* Qh  = (_Float16*)w; w += (size_t)BS * Hdim * 2;      // 16 MB
  _Float16* Kh  = (_Float16*)w; w += (size_t)BS * Hdim * 2;      // 16 MB
  _Float16* Vt  = (_Float16*)w; w += (size_t)BS * Hdim * 2;      // 16 MB

  cvt_f32_f16<<<4096, 256, 0, stream>>>(X, Xh, BS * Hdim);
  cvt_w_t<<<4096, 256, 0, stream>>>(Wq, Wtq);
  cvt_w_t<<<4096, 256, 0, stream>>>(Wk, Wtk);
  cvt_w_t<<<4096, 256, 0, stream>>>(Wv, Wtv);

  // Fold 1/sqrt(HD) and log2(e) into Q so softmax uses pure exp2.
  const float QSCALE = 0.125f * 1.4426950408889634f;
  qkv_proj<<<1024, 256, 0, stream>>>(Xh, Wtq, bq, Qh, QSCALE, 0);
  qkv_proj<<<1024, 256, 0, stream>>>(Xh, Wtk, bk, Kh, 1.0f, 0);
  qkv_proj<<<1024, 256, 0, stream>>>(Xh, Wtv, bv, Vt, 1.0f, 1);

  attn_kernel<<<1024, 256, 0, stream>>>(Qh, Kh, Vt, out);
}
